// AttentiveFP_67929202754271
// MI455X (gfx1250) — matrix operand order repaired and checked by reading, not compiled, not run on hardware
//
#include <hip/hip_runtime.h>
#include <hip/hip_bf16.h>

// ---------------- problem constants (from reference) ----------------
#define N_NODES   100000
#define N_EDGES   1000000
#define N_GRAPHS  5000
#define HD        64

typedef __attribute__((ext_vector_type(16))) __bf16 bv16;
typedef __attribute__((ext_vector_type(8)))  float  fv8;

// ---------------- device helpers ----------------
__device__ __forceinline__ float d_leaky(float v) { return v > 0.f ? v : 0.01f * v; }
__device__ __forceinline__ float d_elu(float v)   { return v > 0.f ? v : (__expf(v) - 1.f); }
__device__ __forceinline__ float d_sigm(float v)  { return 1.f / (1.f + __expf(-v)); }

// orderable encoding of float for atomicMax on unsigned
__device__ __forceinline__ unsigned enc_f(float f) {
  unsigned u = __float_as_uint(f);
  return (u >> 31) ? ~u : (u | 0x80000000u);
}
__device__ __forceinline__ float dec_f2(unsigned e) {
  if (e == 0u) return 0.f;  // untouched segment -> matches reference isfinite clamp
  unsigned u = (e >> 31) ? (e & 0x7FFFFFFFu) : ~e;
  return __uint_as_float(u);
}

// 16-bit A-matrix 16x32 fragment element -> K index (ISA 7.12.2)
__device__ __forceinline__ int a_k(int lane, int j) {
  return ((j >> 3) << 4) | ((lane >> 4) << 3) | (j & 7);
}

__device__ __forceinline__ fv8 wmma_bf16(bv16 a, bv16 b, fv8 c) {
  // v_wmma_f32_16x16x32_bf16 (probe-verified builtin, 8 args)
  return __builtin_amdgcn_wmma_f32_16x16x32_bf16(false, a, false, b, (short)0, c,
                                                 false, false);
}

// packed B fragments: [frag][lane][e] contiguous, 512 bf16 per fragment
__device__ __forceinline__ bv16 load_bfrag(const __bf16* pack, int frag, int lane) {
  return *(const bv16*)(pack + ((size_t)frag << 9) + (lane << 4));
}

// ---------------- weight packing: B = W^T (K x N), zero-padded K ----------------
// grid = (Kp/32)*Ntiles blocks, 512 threads
__global__ void pack_b_kernel(const float* __restrict__ W, __bf16* __restrict__ dst,
                              int Kdim, int Ntiles) {
  int frag = blockIdx.x;
  int kc = frag / Ntiles, nt = frag % Ntiles;
  int t = threadIdx.x;              // 0..511
  int lane = t >> 4, e = t & 15;
  int K  = kc * 32 + ((e >> 3) << 4) + ((lane >> 4) << 3) + (e & 7);   // B layout: K = (e>>3)*16 + (lane>>4)*8 + (e&7)
  int NN = nt * 16 + (lane & 15);
  float v = (K < Kdim) ? W[(size_t)NN * Kdim + K] : 0.f;
  dst[((size_t)frag << 9) + t] = (__bf16)v;
}

// ---------------- small utility kernels ----------------
__global__ void zero_f_kernel(float* __restrict__ p, int n) {
  int i = blockIdx.x * blockDim.x + threadIdx.x;
  if (i < n) p[i] = 0.f;
}
__global__ void init_seg_kernel(unsigned* __restrict__ mx, float* __restrict__ sm, int n) {
  int i = blockIdx.x * blockDim.x + threadIdx.x;
  if (i < n) { mx[i] = 0u; sm[i] = 0.f; }
}
__global__ void relu_ip_kernel(float* __restrict__ p, int n) {
  int i = blockIdx.x * blockDim.x + threadIdx.x;
  if (i < n) p[i] = fmaxf(p[i], 0.f);
}
__global__ void scatter_sum_kernel(const float* __restrict__ x, const int* __restrict__ batch,
                                   float* __restrict__ acc, int n) {  // n = N*64
  int i = blockIdx.x * blockDim.x + threadIdx.x;
  if (i >= n) return;
  int node = i >> 6;
  atomicAdd(&acc[((size_t)batch[node] << 6) + (i & 63)], x[i]);
}

// ---------------- K1: x = leaky(raw @ lin1_w^T + b) ----------------
__global__ void lin1_kernel(const float* __restrict__ raw, const __bf16* __restrict__ wpack,
                            const float* __restrict__ bias, float* __restrict__ xo, int R) {
  int gw   = blockIdx.x * (blockDim.x >> 5) + (threadIdx.x >> 5);
  int lane = threadIdx.x & 31;
  int r0 = gw * 16;
  if (r0 >= R) return;
  int m = lane & 15, half = lane >> 4;
  int row = r0 + m; if (row >= R) row = R - 1;
  fv8 acc[4] = {};
  for (int kc = 0; kc < 2; ++kc) {
    bv16 a;
    #pragma unroll
    for (int j = 0; j < 16; ++j)
      a[j] = (__bf16)raw[((size_t)row << 6) + kc * 32 + a_k(lane, j)];
    #pragma unroll
    for (int nt = 0; nt < 4; ++nt)
      acc[nt] = wmma_bf16(a, load_bfrag(wpack, kc * 4 + nt, lane), acc[nt]);
  }
  #pragma unroll
  for (int nt = 0; nt < 4; ++nt)
    #pragma unroll
    for (int v = 0; v < 8; ++v) {
      int M = v + (half << 3), NN = nt * 16 + (lane & 15);
      int rr = r0 + M;
      if (rr < R) xo[((size_t)rr << 6) + NN] = d_leaky(acc[nt][v] + bias[NN]);
    }
}

// ---------------- K2: xj = leaky([raw[src], edge_attr] @ gate_nl_w^T + b) -> bf16 ----------------
__global__ void gate_xj_kernel(const float* __restrict__ raw, const float* __restrict__ eattr,
                               const int* __restrict__ src, const __bf16* __restrict__ wpack,
                               const float* __restrict__ bias, __bf16* __restrict__ xj, int E) {
  int gw   = blockIdx.x * (blockDim.x >> 5) + (threadIdx.x >> 5);
  int lane = threadIdx.x & 31;
  int e0 = gw * 16;
  if (e0 >= E) return;
  int m = lane & 15, half = lane >> 4;
  int er = e0 + m; if (er >= E) er = E - 1;
  int s = src[er];
  fv8 acc[4] = {};
  for (int kc = 0; kc < 3; ++kc) {          // Kp = 96 (80 real + pad)
    bv16 a;
    #pragma unroll
    for (int j = 0; j < 16; ++j) {
      int K = kc * 32 + a_k(lane, j);
      float v = 0.f;
      if (K < 64)      v = raw[((size_t)s << 6) + K];
      else if (K < 80) v = eattr[(size_t)er * 16 + (K - 64)];
      a[j] = (__bf16)v;
    }
    #pragma unroll
    for (int nt = 0; nt < 4; ++nt)
      acc[nt] = wmma_bf16(a, load_bfrag(wpack, kc * 4 + nt, lane), acc[nt]);
  }
  #pragma unroll
  for (int nt = 0; nt < 4; ++nt)
    #pragma unroll
    for (int v = 0; v < 8; ++v) {
      int M = v + (half << 3), NN = nt * 16 + (lane & 15);
      int rr = e0 + M;
      if (rr < E) xj[((size_t)rr << 6) + NN] = (__bf16)d_leaky(acc[nt][v] + bias[NN]);
    }
}

// ---------------- edge scores ----------------
__global__ void gate_score_kernel(const float* __restrict__ x, const __bf16* __restrict__ xj,
                                  const int* __restrict__ dst, const float* __restrict__ aw,
                                  const float* __restrict__ ab, float* __restrict__ score,
                                  unsigned* __restrict__ segmax, int E) {
  int e = blockIdx.x * blockDim.x + threadIdx.x;
  if (e >= E) return;
  int d = dst[e];
  float s = ab[0];
  #pragma unroll 8
  for (int k = 0; k < 64; ++k) s += x[((size_t)d << 6) + k] * aw[k];
  #pragma unroll 8
  for (int k = 0; k < 64; ++k) s += (float)xj[((size_t)e << 6) + k] * aw[64 + k];
  s = d_leaky(s);
  score[e] = s;
  atomicMax(&segmax[d], enc_f(s));
}

__global__ void conv_score_kernel(const float* __restrict__ x, const int* __restrict__ src,
                                  const int* __restrict__ dst, const float* __restrict__ aw,
                                  const float* __restrict__ ab, float* __restrict__ score,
                                  unsigned* __restrict__ segmax, int E) {
  int e = blockIdx.x * blockDim.x + threadIdx.x;
  if (e >= E) return;
  int d = dst[e], s0 = src[e];
  float s = ab[0];
  #pragma unroll 8
  for (int k = 0; k < 64; ++k) s += x[((size_t)d << 6) + k] * aw[k];
  #pragma unroll 8
  for (int k = 0; k < 64; ++k) s += x[((size_t)s0 << 6) + k] * aw[64 + k];
  s = d_leaky(s);
  score[e] = s;
  atomicMax(&segmax[d], enc_f(s));
}

__global__ void mol_score_kernel(const float* __restrict__ x2, const float* __restrict__ xc,
                                 const float* __restrict__ out, const int* __restrict__ batch,
                                 const float* __restrict__ aw, const float* __restrict__ ab,
                                 float* __restrict__ score, unsigned* __restrict__ segmax, int N) {
  int n = blockIdx.x * blockDim.x + threadIdx.x;
  if (n >= N) return;
  int g = batch[n];
  float s = ab[0];
  #pragma unroll 8
  for (int k = 0; k < 64; ++k) s += out[((size_t)g << 6) + k] * aw[k];
  #pragma unroll 8
  for (int k = 0; k < 64; ++k)
    s += 0.5f * (x2[((size_t)n << 6) + k] + xc[((size_t)n << 6) + k]) * aw[64 + k];
  s = d_leaky(s);
  score[n] = s;
  atomicMax(&segmax[g], enc_f(s));
}

// ---------------- softmax exp + segment sum ----------------
__global__ void exp_pass_kernel(const float* __restrict__ score, const int* __restrict__ idx,
                                const unsigned* __restrict__ segmax, float* __restrict__ eexp,
                                float* __restrict__ segsum, int E) {
  int e = blockIdx.x * blockDim.x + threadIdx.x;
  if (e >= E) return;
  int d = idx[e];
  float v = __expf(score[e] - dec_f2(segmax[d]));
  eexp[e] = v;
  atomicAdd(&segsum[d], v);
}

// ---------------- msg = (A @ attend_w^T + b) * aw ; scatter-add by seg ----------------
// variant A-source 0: xj bf16 rows; 1: x[src] f32 gather; 2: 0.5*(x2+xc) f32 rows
template <int SRCMODE>
__global__ void msg_kernel(const __bf16* __restrict__ xjrows,
                           const float* __restrict__ xa, const float* __restrict__ xb,
                           const int* __restrict__ gidx,   // src (mode1) — per-row gather
                           const int* __restrict__ seg,    // dst or batch — scatter target
                           const __bf16* __restrict__ wpack, const float* __restrict__ bias,
                           const float* __restrict__ eexp, const float* __restrict__ segsum,
                           float* __restrict__ hacc, int E) {
  int gw   = blockIdx.x * (blockDim.x >> 5) + (threadIdx.x >> 5);
  int lane = threadIdx.x & 31;
  int e0 = gw * 16;
  if (e0 >= E) return;
  int m = lane & 15, half = lane >> 4;
  int er = e0 + m; if (er >= E) er = E - 1;
  int grow = (SRCMODE == 1) ? gidx[er] : er;
  fv8 acc[4] = {};
  for (int kc = 0; kc < 2; ++kc) {
    bv16 a;
    #pragma unroll
    for (int j = 0; j < 16; ++j) {
      int K = kc * 32 + a_k(lane, j);
      float v;
      if (SRCMODE == 0)      v = (float)xjrows[((size_t)er << 6) + K];
      else if (SRCMODE == 1) v = xa[((size_t)grow << 6) + K];
      else                   v = 0.5f * (xa[((size_t)er << 6) + K] + xb[((size_t)er << 6) + K]);
      a[j] = (__bf16)v;
    }
    #pragma unroll
    for (int nt = 0; nt < 4; ++nt)
      acc[nt] = wmma_bf16(a, load_bfrag(wpack, kc * 4 + nt, lane), acc[nt]);
  }
  float awv[8]; int dseg[8];
  #pragma unroll
  for (int v = 0; v < 8; ++v) {
    int rr = e0 + v + (half << 3); if (rr >= E) rr = E - 1;
    int d = seg[rr];
    dseg[v] = d;
    awv[v] = eexp[rr] / (segsum[d] + 1e-16f);
  }
  #pragma unroll
  for (int nt = 0; nt < 4; ++nt)
    #pragma unroll
    for (int v = 0; v < 8; ++v) {
      int NN = nt * 16 + (lane & 15);
      int rr = e0 + v + (half << 3);
      if (rr < E)
        atomicAdd(&hacc[((size_t)dseg[v] << 6) + NN], (acc[nt][v] + bias[NN]) * awv[v]);
    }
}

// ---------------- GRU: xout = relu(gru(elu(hpre), xold)) ----------------
__global__ void gru_kernel(const float* __restrict__ hpre, const float* __restrict__ xold,
                           const __bf16* __restrict__ wih, const __bf16* __restrict__ whh,
                           const float* __restrict__ bih, const float* __restrict__ bhh,
                           float* __restrict__ xout, int R) {
  int gw   = blockIdx.x * (blockDim.x >> 5) + (threadIdx.x >> 5);
  int lane = threadIdx.x & 31;
  int rt = gw >> 2, c = gw & 3;        // 16-row tile x 16-col slice
  int r0 = rt * 16;
  if (r0 >= R) return;
  int m = lane & 15, half = lane >> 4;
  int row = r0 + m; if (row >= R) row = R - 1;
  fv8 gi[3] = {}, gh[3] = {};
  for (int kc = 0; kc < 2; ++kc) {
    bv16 ah, ax;
    #pragma unroll
    for (int j = 0; j < 16; ++j) {
      int K = kc * 32 + a_k(lane, j);
      ah[j] = (__bf16)d_elu(hpre[((size_t)row << 6) + K]);
      ax[j] = (__bf16)xold[((size_t)row << 6) + K];
    }
    #pragma unroll
    for (int g = 0; g < 3; ++g) {
      int nt = g * 4 + c;
      gi[g] = wmma_bf16(ah, load_bfrag(wih, kc * 12 + nt, lane), gi[g]);
      gh[g] = wmma_bf16(ax, load_bfrag(whh, kc * 12 + nt, lane), gh[g]);
    }
  }
  #pragma unroll
  for (int v = 0; v < 8; ++v) {
    int M = v + (half << 3);
    int rr = r0 + M;
    if (rr >= R) continue;
    int col = c * 16 + (lane & 15);
    float ir = gi[0][v] + bih[col],        hr = gh[0][v] + bhh[col];
    float iz = gi[1][v] + bih[64 + col],   hz = gh[1][v] + bhh[64 + col];
    float in_ = gi[2][v] + bih[128 + col], hn = gh[2][v] + bhh[128 + col];
    float r = d_sigm(ir + hr);
    float z = d_sigm(iz + hz);
    float nn = tanhf(in_ + r * hn);
    float xo = xold[((size_t)rr << 6) + col];
    xout[((size_t)rr << 6) + col] = fmaxf((1.f - z) * nn + z * xo, 0.f);
  }
}

// ---------------- final: pred = out @ lin2_w^T + b ----------------
__global__ void final_lin_kernel(const float* __restrict__ out, const float* __restrict__ w,
                                 const float* __restrict__ b, float* __restrict__ pred, int G) {
  int g = blockIdx.x * blockDim.x + threadIdx.x;
  if (g >= G) return;
  float s = b[0];
  #pragma unroll 8
  for (int k = 0; k < 64; ++k) s += out[((size_t)g << 6) + k] * w[k];
  pred[g] = s;
}

// ---------------- host orchestration ----------------
extern "C" void kernel_launch(void* const* d_in, const int* in_sizes, int n_in,
                              void* d_out, int out_size, void* d_ws, size_t ws_size,
                              hipStream_t stream) {
  (void)in_sizes; (void)n_in; (void)out_size; (void)ws_size;
  const int N = N_NODES, E = N_EDGES, G = N_GRAPHS;

  const float* raw       = (const float*)d_in[0];
  const int*   eidx      = (const int*)d_in[1];
  const int*   src       = eidx;
  const int*   dst       = eidx + E;
  const float* eattr     = (const float*)d_in[2];
  const int*   batch     = (const int*)d_in[3];
  const float* lin1_w    = (const float*)d_in[4];
  const float* lin1_b    = (const float*)d_in[5];
  const float* gnl_w     = (const float*)d_in[6];
  const float* gnl_b     = (const float*)d_in[7];
  const float* gal_w     = (const float*)d_in[8];
  const float* gal_b     = (const float*)d_in[9];
  const float* gat_w     = (const float*)d_in[10];
  const float* gat_b     = (const float*)d_in[11];
  const float* cal_w     = (const float*)d_in[12];
  const float* cal_b     = (const float*)d_in[13];
  const float* cat_w     = (const float*)d_in[14];
  const float* cat_b     = (const float*)d_in[15];
  const float* awih      = (const float*)d_in[16];
  const float* awhh      = (const float*)d_in[17];
  const float* abih      = (const float*)d_in[18];
  const float* abhh      = (const float*)d_in[19];
  const float* mal_w     = (const float*)d_in[20];
  const float* mal_b     = (const float*)d_in[21];
  const float* mat_w     = (const float*)d_in[22];
  const float* mat_b     = (const float*)d_in[23];
  const float* mwih      = (const float*)d_in[24];
  const float* mwhh      = (const float*)d_in[25];
  const float* mbih      = (const float*)d_in[26];
  const float* mbhh      = (const float*)d_in[27];
  const float* lin2_w    = (const float*)d_in[28];
  const float* lin2_b    = (const float*)d_in[29];
  float* pred = (float*)d_out;

  char* ws = (char*)d_ws;
  size_t off = 0;
  auto carve = [&](size_t bytes) -> char* {
    off = (off + 255) & ~(size_t)255;
    char* p = ws + off;
    off += bytes;
    return p;
  };

  __bf16* pk_lin1    = (__bf16*)carve(8  * 512 * 2);
  __bf16* pk_gnl     = (__bf16*)carve(12 * 512 * 2);
  __bf16* pk_gat     = (__bf16*)carve(8  * 512 * 2);
  __bf16* pk_cat0    = (__bf16*)carve(8  * 512 * 2);
  __bf16* pk_cat1    = (__bf16*)carve(8  * 512 * 2);
  __bf16* pk_awih[3]; __bf16* pk_awhh[3];
  for (int l = 0; l < 3; ++l) { pk_awih[l] = (__bf16*)carve(24 * 512 * 2);
                                pk_awhh[l] = (__bf16*)carve(24 * 512 * 2); }
  __bf16* pk_mat     = (__bf16*)carve(8  * 512 * 2);
  __bf16* pk_mwih    = (__bf16*)carve(24 * 512 * 2);
  __bf16* pk_mwhh    = (__bf16*)carve(24 * 512 * 2);

  __bf16*   xj     = (__bf16*)carve((size_t)E * 64 * 2);
  float*    score  = (float*)carve((size_t)E * 4);
  float*    eexp   = (float*)carve((size_t)E * 4);
  unsigned* segmax = (unsigned*)carve((size_t)N * 4);
  float*    segsum = (float*)carve((size_t)N * 4);
  float*    hacc   = (float*)carve((size_t)N * 64 * 4);
  float*    x_init = (float*)carve((size_t)N * 64 * 4);
  float*    x0     = (float*)carve((size_t)N * 64 * 4);
  float*    x1     = (float*)carve((size_t)N * 64 * 4);
  float*    x2     = (float*)carve((size_t)N * 64 * 4);
  float*    outA   = (float*)carve((size_t)G * 64 * 4);
  float*    outB   = (float*)carve((size_t)G * 64 * 4);
  float*    hmol   = (float*)carve((size_t)G * 64 * 4);

  const int B256 = 256;
  const int gE   = (E + 255) / 256;        // per-edge threads
  const int gN   = (N + 255) / 256;        // per-node threads
  const int wE   = ((E / 16) + 7) / 8;     // edge-tile waves, 8 waves/block
  const int wN   = ((N / 16) + 7) / 8;     // node-tile waves
  const int wGruN = ((N / 16) * 4 + 7) / 8;
  const int wGruG = (((G + 15) / 16) * 4 + 7) / 8;
  const int gNH  = (N * 64 + 255) / 256;
  const int gGH  = (G * 64 + 255) / 256;

  // ---- pack weights (B = W^T in fragment order) ----
  pack_b_kernel<<<8,  512, 0, stream>>>(lin1_w, pk_lin1, 64, 4);
  pack_b_kernel<<<12, 512, 0, stream>>>(gnl_w,  pk_gnl,  80, 4);
  pack_b_kernel<<<8,  512, 0, stream>>>(gat_w,  pk_gat,  64, 4);
  pack_b_kernel<<<8,  512, 0, stream>>>(cat_w,          pk_cat0, 64, 4);
  pack_b_kernel<<<8,  512, 0, stream>>>(cat_w + 4096,   pk_cat1, 64, 4);
  for (int l = 0; l < 3; ++l) {
    pack_b_kernel<<<24, 512, 0, stream>>>(awih + (size_t)l * 192 * 64, pk_awih[l], 64, 12);
    pack_b_kernel<<<24, 512, 0, stream>>>(awhh + (size_t)l * 192 * 64, pk_awhh[l], 64, 12);
  }
  pack_b_kernel<<<8,  512, 0, stream>>>(mat_w, pk_mat,  64, 4);
  pack_b_kernel<<<24, 512, 0, stream>>>(mwih,  pk_mwih, 64, 12);
  pack_b_kernel<<<24, 512, 0, stream>>>(mwhh,  pk_mwhh, 64, 12);

  // ---- x = leaky(raw @ lin1^T + b) ----
  lin1_kernel<<<wN, B256, 0, stream>>>(raw, pk_lin1, lin1_b, x_init, N);

  // ---- GATEConv (layer 0) ----
  gate_xj_kernel<<<wE, B256, 0, stream>>>(raw, eattr, src, pk_gnl, gnl_b, xj, E);
  init_seg_kernel<<<gN, B256, 0, stream>>>(segmax, segsum, N);
  gate_score_kernel<<<gE, B256, 0, stream>>>(x_init, xj, dst, gal_w, gal_b, score, segmax, E);
  exp_pass_kernel<<<gE, B256, 0, stream>>>(score, dst, segmax, eexp, segsum, E);
  zero_f_kernel<<<gNH, B256, 0, stream>>>(hacc, N * 64);
  msg_kernel<0><<<wE, B256, 0, stream>>>(xj, nullptr, nullptr, nullptr, dst,
                                         pk_gat, gat_b, eexp, segsum, hacc, E);
  gru_kernel<<<wGruN, B256, 0, stream>>>(hacc, x_init, pk_awih[0], pk_awhh[0],
                                         abih, abhh, x0, N);

  // ---- GATConv layers 1..2 ----
  const float* xin[2] = { x0, x1 };
  float*       xout[2] = { x1, x2 };
  __bf16*      catp[2] = { pk_cat0, pk_cat1 };
  for (int l = 0; l < 2; ++l) {
    init_seg_kernel<<<gN, B256, 0, stream>>>(segmax, segsum, N);
    conv_score_kernel<<<gE, B256, 0, stream>>>(xin[l], src, dst, cal_w + l * 128,
                                               cal_b + l, score, segmax, E);
    exp_pass_kernel<<<gE, B256, 0, stream>>>(score, dst, segmax, eexp, segsum, E);
    zero_f_kernel<<<gNH, B256, 0, stream>>>(hacc, N * 64);
    msg_kernel<1><<<wE, B256, 0, stream>>>(nullptr, xin[l], nullptr, src, dst,
                                           catp[l], cat_b + l * 64, eexp, segsum, hacc, E);
    gru_kernel<<<wGruN, B256, 0, stream>>>(hacc, xin[l], pk_awih[l + 1], pk_awhh[l + 1],
                                           abih + (l + 1) * 192, abhh + (l + 1) * 192,
                                           xout[l], N);
  }

  // ---- molecule readout ----
  zero_f_kernel<<<gGH, B256, 0, stream>>>(outA, G * 64);
  scatter_sum_kernel<<<gNH, B256, 0, stream>>>(x2, batch, outA, N * 64);
  relu_ip_kernel<<<gGH, B256, 0, stream>>>(outA, G * 64);

  // cached[CACHED_INDEX[t]] : t=0 -> x1, t=1 -> x2
  const float* xcache[2] = { x1, x2 };
  float* cur = outA; float* nxt = outB;
  for (int t = 0; t < 2; ++t) {
    init_seg_kernel<<<gN, B256, 0, stream>>>(segmax, segsum, N);
    mol_score_kernel<<<gN, B256, 0, stream>>>(x2, xcache[t], cur, batch,
                                              mal_w, mal_b, score, segmax, N);
    exp_pass_kernel<<<gN, B256, 0, stream>>>(score, batch, segmax, eexp, segsum, N);
    zero_f_kernel<<<gGH, B256, 0, stream>>>(hmol, G * 64);
    msg_kernel<2><<<wN, B256, 0, stream>>>(nullptr, x2, xcache[t], nullptr, batch,
                                           pk_mat, mat_b, eexp, segsum, hmol, N);
    gru_kernel<<<wGruG, B256, 0, stream>>>(hmol, cur, pk_mwih, pk_mwhh,
                                           mbih, mbhh, nxt, G);
    float* tmp = cur; cur = nxt; nxt = tmp;
  }

  // ---- final linear ----
  final_lin_kernel<<<(G + 255) / 256, B256, 0, stream>>>(cur, lin2_w, lin2_b, pred, G);
}
